// BlockLayer_23373212025096
// MI455X (gfx1250) — matrix operand order repaired and checked by reading, not compiled, not run on hardware
//
#include <hip/hip_runtime.h>
#include <hip/hip_bf16.h>

// ---------------------------------------------------------------------------
// Transformer block (causal MHA + LN residual + top-2 MoE + LN residual)
// MI455X / gfx1250: wave32, bf16 WMMA, bf16-converted weights (134MB -> fits
// the 192MB L2), wide b128 fragment loads, async global->LDS staging.
// ---------------------------------------------------------------------------

#define BB   2
#define TT   1024
#define DD   1024
#define HH   16
#define HS   64
#define EE   8
#define FFD  4096
#define NTOK (BB * TT)     // 2048 tokens

typedef __bf16 bf16_t;
typedef __attribute__((ext_vector_type(16))) __bf16 v16bf;
typedef __attribute__((ext_vector_type(8)))  __bf16 v8bf;
typedef __attribute__((ext_vector_type(4)))  __bf16 v4bf;
typedef __attribute__((ext_vector_type(4)))  float  v4f;
typedef __attribute__((ext_vector_type(8)))  float  v8f;
typedef __attribute__((ext_vector_type(4)))  int    v4i;

#define AS1 __attribute__((address_space(1)))
#define AS3 __attribute__((address_space(3)))

#if defined(__has_builtin)
#if __has_builtin(__builtin_amdgcn_global_load_async_to_lds_b128) && \
    __has_builtin(__builtin_amdgcn_s_wait_asynccnt)
#define HAS_ASYNC_LDS 1
#endif
#endif

// integer round-trips avoid address-space cast legality issues; generic LDS
// pointer's low 32 bits are the LDS byte address (ISA 10.2)
__device__ __forceinline__ AS1 v4i* gptr_v4i(const void* p) {
  return (AS1 v4i*)(uintptr_t)p;
}
__device__ __forceinline__ AS3 v4i* lptr_v4i(const void* p) {
  return (AS3 v4i*)(uint32_t)(uintptr_t)p;
}

// ---------------------------------------------------------------------------
// WMMA fragment helpers (wave32). A/B k order below is this code's contiguous choice, not the ISA table's.
// A (16x32 bf16): lane m=L&15, koff=16*(L>>4); per lane k = koff..koff+15
//                 -> one contiguous 32B run of a row-major bf16 row.
// B (32x16 bf16): lane n=L&15, kb=16*(L>>4); per lane k = kb..kb+15 -> one
//                 contiguous 32B run of a *transposed* (n-major) source.
// C/D (16x16 f32): VGPR r -> row r (lanes 0-15) / r+8 (lanes 16-31), col=L&15.
// ---------------------------------------------------------------------------

__device__ __forceinline__ v8f wmma_bf16(v16bf a, v16bf b, v8f c) {
  return __builtin_amdgcn_wmma_f32_16x16x32_bf16(
      false, a, false, b, (short)0, c, false, false);
}

// A fragment from row-major bf16: element (m,k) = src[m*ld + k]
__device__ __forceinline__ v16bf load_a_bf16(const bf16_t* src, int ld) {
  const int lane = threadIdx.x & 31;
  const int m = lane & 15;
  const int koff = (lane >> 4) << 4;
  const bf16_t* p = src + m * ld + koff;
  v8bf lo = *(const v8bf*)p;          // k = koff .. koff+7
  v8bf hi = *(const v8bf*)(p + 8);    // k = koff+8 .. koff+15
  v16bf a;
#pragma unroll
  for (int e = 0; e < 8; ++e) { a[e] = lo[e]; a[e + 8] = hi[e]; }
  return a;
}

// B fragment from n-major (pre-transposed) bf16: element (k,n) = src[n*ld + k]
__device__ __forceinline__ v16bf load_bt_bf16(const bf16_t* src, int ld) {
  const int lane = threadIdx.x & 31;
  const int n = lane & 15;
  const int kb = (lane >> 4) << 4;
  const bf16_t* p = src + n * ld + kb;
  v8bf lo = *(const v8bf*)p;
  v8bf hi = *(const v8bf*)(p + 8);
  v16bf b;
#pragma unroll
  for (int e = 0; e < 8; ++e) { b[e] = lo[e]; b[e + 8] = hi[e]; }
  return b;
}

__device__ __forceinline__ v8bf vzero8() {
  v8bf z;
#pragma unroll
  for (int e = 0; e < 8; ++e) z[e] = (bf16_t)0.f;
  return z;
}

// cross-lane reductions
__device__ __forceinline__ float hredmax(float x) {
#pragma unroll
  for (int off = 1; off < 16; off <<= 1) x = fmaxf(x, __shfl_xor(x, off, 32));
  return x;
}
__device__ __forceinline__ float hredsum(float x) {
#pragma unroll
  for (int off = 1; off < 16; off <<= 1) x += __shfl_xor(x, off, 32);
  return x;
}
__device__ __forceinline__ float wredsum(float x) {
#pragma unroll
  for (int off = 16; off >= 1; off >>= 1) x += __shfl_xor(x, off, 32);
  return x;
}

// ---------------------------------------------------------------------------
// Prep kernel A: f32 -> bf16 elementwise (vectorized 4-wide)
// ---------------------------------------------------------------------------
__global__ __launch_bounds__(256) void cvt_bf16_kernel(
    const float* __restrict__ src, bf16_t* __restrict__ dst, int n4) {
  int i = (blockIdx.x * 256 + threadIdx.x);
  if (i < n4) {
    v4f f = *(const v4f*)(src + (size_t)i * 4);
    v4bf o;
#pragma unroll
    for (int e = 0; e < 4; ++e) o[e] = (bf16_t)f[e];
    *(v4bf*)(dst + (size_t)i * 4) = o;
  }
}

// ---------------------------------------------------------------------------
// Prep kernel B: batched transpose + convert.  src[z][r][c] -> dst[z][c][r].
// 32x32 LDS tiles, block = 256 (32x8).
// ---------------------------------------------------------------------------
__global__ __launch_bounds__(256) void tcvt_kernel(
    const float* __restrict__ src, bf16_t* __restrict__ dst, int R, int C) {
  __shared__ float tile[32][33];
  const size_t base = (size_t)blockIdx.z * R * C;
  const int r0 = blockIdx.x * 32, c0 = blockIdx.y * 32;
  const int tr = threadIdx.x & 31, tc = threadIdx.x >> 5;
#pragma unroll
  for (int i = 0; i < 4; ++i)
    tile[tc + i * 8][tr] = src[base + (size_t)(r0 + tc + i * 8) * C + c0 + tr];
  __syncthreads();
#pragma unroll
  for (int i = 0; i < 4; ++i)
    dst[base + (size_t)(c0 + tc + i * 8) * R + r0 + tr] =
        (bf16_t)tile[tr][tc + i * 8];
}

// ---------------------------------------------------------------------------
// Kernel 1: Q/K/V projection.  grid=(N/64, H, 3), block=128 (4 waves).
// A = xbf [N,D] bf16; B = Wt[h] (pre-transposed, [HS,D] n-major).
// q,k stored bf16 [B,H,T,HS]; v stored bf16 TRANSPOSED [B,H,HS,T] so the
// attention P.V operand is contiguous.
// ---------------------------------------------------------------------------
__global__ __launch_bounds__(128) void qkv_kernel(
    const bf16_t* __restrict__ xbf, const bf16_t* __restrict__ Wqt,
    const bf16_t* __restrict__ Wkt, const bf16_t* __restrict__ Wvt,
    bf16_t* __restrict__ q, bf16_t* __restrict__ k, bf16_t* __restrict__ v) {
  const int wave = threadIdx.x >> 5;
  const int lane = threadIdx.x & 31;
  const int h = blockIdx.y;
  const int which = blockIdx.z;
  const bf16_t* Wt = (which == 0) ? Wqt : (which == 1) ? Wkt : Wvt;
  bf16_t* out = (which == 0) ? q : (which == 1) ? k : v;
  const int m0 = blockIdx.x * 64 + wave * 16;
  const bf16_t* Wth = Wt + (size_t)h * HS * DD;   // [HS, D] n-major

  v8f acc[4] = {};
  for (int kk = 0; kk < DD; kk += 32) {
    __builtin_prefetch(Wth + kk + 64, 0, 1);
    v16bf a = load_a_bf16(xbf + (size_t)m0 * DD + kk, DD);
#pragma unroll
    for (int j = 0; j < 4; ++j) {
      v16bf b = load_bt_bf16(Wth + (size_t)(j * 16) * DD + kk, DD);
      acc[j] = wmma_bf16(a, b, acc[j]);
    }
  }
  const int n = lane & 15;
  const int mb = (lane >> 4) << 3;
#pragma unroll
  for (int j = 0; j < 4; ++j)
#pragma unroll
    for (int r = 0; r < 8; ++r) {
      int m = m0 + mb + r;
      int b_ = m / TT, t = m % TT;
      int f = j * 16 + n;
      if (which == 2)  // v transposed: [B,H,HS,T]
        out[(((size_t)b_ * HH + h) * HS + f) * TT + t] = (bf16_t)acc[j][r];
      else
        out[(((size_t)b_ * HH + h) * TT + t) * HS + f] = (bf16_t)acc[j][r];
    }
}

// ---------------------------------------------------------------------------
// Kernel 2: flash-style causal attention.  grid=(T/64, B*H), block=128.
// Per wave: 16-query tile, 32-key chunks, online softmax.  P transposed
// through per-wave bf16 LDS staging.  Scale = D**-0.5 (reference quirk).
// ---------------------------------------------------------------------------
__global__ __launch_bounds__(128) void attn_kernel(
    const bf16_t* __restrict__ q, const bf16_t* __restrict__ k,
    const bf16_t* __restrict__ vt, float* __restrict__ attn_out) {
  __shared__ bf16_t lds_p[4][16 * 32];
  const int wave = threadIdx.x >> 5;
  const int lane = threadIdx.x & 31;
  const int bh = blockIdx.y;
  const int b_ = bh / HH, h = bh % HH;
  const int m0 = (blockIdx.x * 4 + wave) * 16;
  const bf16_t* qp = q + (size_t)bh * TT * HS;
  const bf16_t* kp = k + (size_t)bh * TT * HS;
  const bf16_t* vp = vt + (size_t)bh * HS * TT;   // [HS, T]
  bf16_t* pbuf = lds_p[wave];

  v16bf qa[2];
  qa[0] = load_a_bf16(qp + (size_t)m0 * HS, HS);
  qa[1] = load_a_bf16(qp + (size_t)m0 * HS + 32, HS);

  const float scale = rsqrtf((float)DD);
  const int n = lane & 15;
  const int mb = (lane >> 4) << 3;

  v8f oacc[4] = {};
  float rmax[8], rsum[8];
#pragma unroll
  for (int r = 0; r < 8; ++r) { rmax[r] = -1e30f; rsum[r] = 0.f; }

  for (int s0 = 0; s0 <= m0; s0 += 32) {
    v8f s[2] = {};
#pragma unroll
    for (int j = 0; j < 2; ++j) {
      s[j] = wmma_bf16(qa[0], load_bt_bf16(kp + (size_t)(s0 + 16 * j) * HS, HS), s[j]);
      s[j] = wmma_bf16(qa[1], load_bt_bf16(kp + (size_t)(s0 + 16 * j) * HS + 32, HS), s[j]);
    }
#pragma unroll
    for (int r = 0; r < 8; ++r) {
      const int mrow = m0 + mb + r;
#pragma unroll
      for (int j = 0; j < 2; ++j) {
        float val = s[j][r] * scale;
        if (s0 + 16 * j + n > mrow) val = -1e30f;
        s[j][r] = val;
      }
      float mx = hredmax(fmaxf(s[0][r], s[1][r]));
      float nmax = fmaxf(rmax[r], mx);
      float rf = __expf(rmax[r] - nmax);
      rmax[r] = nmax;
      float p0 = __expf(s[0][r] - nmax);
      float p1 = __expf(s[1][r] - nmax);
      rsum[r] = rsum[r] * rf + hredsum(p0 + p1);
      pbuf[(mb + r) * 32 + n]      = (bf16_t)p0;
      pbuf[(mb + r) * 32 + 16 + n] = (bf16_t)p1;
#pragma unroll
      for (int jj = 0; jj < 4; ++jj) oacc[jj][r] *= rf;
    }
    // O += P(16x32) . V(32x64); V operand contiguous thanks to [HS,T] layout
    v16bf pa = load_a_bf16(pbuf, 32);
#pragma unroll
    for (int jj = 0; jj < 4; ++jj) {
      v16bf bv = load_bt_bf16(vp + (size_t)(jj * 16) * TT + s0, TT);
      oacc[jj] = wmma_bf16(pa, bv, oacc[jj]);
    }
  }
#pragma unroll
  for (int jj = 0; jj < 4; ++jj)
#pragma unroll
    for (int r = 0; r < 8; ++r) {
      int mrow = m0 + mb + r;
      attn_out[((size_t)b_ * TT + mrow) * DD + h * HS + jj * 16 + n] =
          oacc[jj][r] / rsum[r];
    }
}

// ---------------------------------------------------------------------------
// Kernel 3/7: out = base + LayerNorm(inp)*g + b; optional bf16 mirror copy.
// ---------------------------------------------------------------------------
__global__ __launch_bounds__(256) void add_ln_kernel(
    const float* __restrict__ base, const float* __restrict__ inp,
    const float* __restrict__ g, const float* __restrict__ bparm,
    float* __restrict__ out, bf16_t* __restrict__ out_bf) {
  __shared__ float red[8];
  const size_t row = blockIdx.x;
  const float* ip = inp + row * DD;
  const float* bp = base + row * DD;
  float vals[4];
  float s = 0.f;
#pragma unroll
  for (int i = 0; i < 4; ++i) { vals[i] = ip[threadIdx.x + i * 256]; s += vals[i]; }
  s = wredsum(s);
  if ((threadIdx.x & 31) == 0) red[threadIdx.x >> 5] = s;
  __syncthreads();
  float tot = 0.f;
#pragma unroll
  for (int w = 0; w < 8; ++w) tot += red[w];
  const float mean = tot / (float)DD;
  __syncthreads();
  float d2 = 0.f;
#pragma unroll
  for (int i = 0; i < 4; ++i) { float dv = vals[i] - mean; d2 += dv * dv; }
  d2 = wredsum(d2);
  if ((threadIdx.x & 31) == 0) red[threadIdx.x >> 5] = d2;
  __syncthreads();
  float v2 = 0.f;
#pragma unroll
  for (int w = 0; w < 8; ++w) v2 += red[w];
  const float inv = rsqrtf(v2 / (float)DD + 1e-5f);
#pragma unroll
  for (int i = 0; i < 4; ++i) {
    int c = threadIdx.x + i * 256;
    float o = bp[c] + (vals[i] - mean) * inv * g[c] + bparm[c];
    out[row * DD + c] = o;
    if (out_bf) out_bf[row * DD + c] = (bf16_t)o;
  }
}

// ---------------------------------------------------------------------------
// Kernel 4: MoE gate — logits = y @ gate_W, top-2, pair softmax, atomic
// token bucketing.  grid = N tokens, block = 256.
// ---------------------------------------------------------------------------
__global__ __launch_bounds__(256) void gate_kernel(
    const float* __restrict__ y, const float* __restrict__ gW,
    int* __restrict__ counts, int* __restrict__ tok_list,
    float* __restrict__ wt_list) {
  __shared__ float sl[EE][8];
  const int tok = blockIdx.x;
  const float* yp = y + (size_t)tok * DD;
  float acc[EE] = {};
  for (int d = threadIdx.x; d < DD; d += 256) {
    float xv = yp[d];
#pragma unroll
    for (int e = 0; e < EE; ++e) acc[e] += xv * gW[d * EE + e];
  }
#pragma unroll
  for (int e = 0; e < EE; ++e) {
    float s = wredsum(acc[e]);
    if ((threadIdx.x & 31) == 0) sl[e][threadIdx.x >> 5] = s;
  }
  __syncthreads();
  if (threadIdx.x == 0) {
    float lg[EE];
#pragma unroll
    for (int e = 0; e < EE; ++e) {
      float s = 0.f;
#pragma unroll
      for (int w = 0; w < 8; ++w) s += sl[e][w];
      lg[e] = s;
    }
    int i0 = 0;
#pragma unroll
    for (int e = 1; e < EE; ++e) if (lg[e] > lg[i0]) i0 = e;
    int i1 = (i0 == 0) ? 1 : 0;
#pragma unroll
    for (int e = 0; e < EE; ++e) if (e != i0 && lg[e] > lg[i1]) i1 = e;
    float mx = fmaxf(lg[i0], lg[i1]);
    float e0 = __expf(lg[i0] - mx), e1 = __expf(lg[i1] - mx);
    float inv = 1.f / (e0 + e1);
    int p0 = atomicAdd(&counts[i0], 1);
    tok_list[i0 * NTOK + p0] = tok;  wt_list[i0 * NTOK + p0] = e0 * inv;
    int p1 = atomicAdd(&counts[i1], 1);
    tok_list[i1 * NTOK + p1] = tok;  wt_list[i1 * NTOK + p1] = e1 * inv;
  }
}

// ---------------------------------------------------------------------------
// Kernel 5: grouped MoE expert GEMM.  grid=(128 tiles, 8 experts), block=256.
// Gathered 16-token bf16 tile staged via async global->LDS (gfx1250 path),
// FC1+ReLU hidden tile (16x4096 bf16 = 128KB) lives in the 320KB WGP LDS,
// FC2 + weighted atomic scatter-add.  Weights are bf16 (L2-resident).
// ---------------------------------------------------------------------------
__global__ __launch_bounds__(256) void moe_kernel(
    const bf16_t* __restrict__ ybf,
    const bf16_t* __restrict__ W1t, const float* __restrict__ b1,
    const bf16_t* __restrict__ W2t, const float* __restrict__ b2,
    const int* __restrict__ counts, const int* __restrict__ tok_list,
    const float* __restrict__ wt_list, float* __restrict__ moe) {
  __shared__ bf16_t aT[16 * DD];     // 32 KB gathered input tile
  __shared__ bf16_t hT[16 * FFD];    // 128 KB hidden tile
  __shared__ int   stok[16];
  __shared__ float swt[16];

  const int e = blockIdx.y;
  const int cnt = counts[e];
  const int t0 = blockIdx.x * 16;
  if (t0 >= cnt) return;

  if (threadIdx.x < 16) {
    int idx = t0 + threadIdx.x;
    stok[threadIdx.x] = (idx < cnt) ? tok_list[e * NTOK + idx] : -1;
    swt[threadIdx.x]  = (idx < cnt) ? wt_list[e * NTOK + idx] : 0.f;
  }
  __syncthreads();

  // ---- stage gathered rows (16 x 1024 bf16) into LDS, 16B per op ----
  for (int i = threadIdx.x; i < 16 * (DD / 8); i += 256) {
    int r = i / (DD / 8);
    int c = (i % (DD / 8)) * 8;
    int tk = stok[r];
    if (tk >= 0) {
#if defined(HAS_ASYNC_LDS)
      __builtin_amdgcn_global_load_async_to_lds_b128(
          gptr_v4i(ybf + (size_t)tk * DD + c), lptr_v4i(&aT[r * DD + c]), 0, 0);
#else
      *(v8bf*)&aT[r * DD + c] = *(const v8bf*)(ybf + (size_t)tk * DD + c);
#endif
    } else {
      *(v8bf*)&aT[r * DD + c] = vzero8();
    }
  }
#if defined(HAS_ASYNC_LDS)
  __builtin_amdgcn_s_wait_asynccnt(0);
#endif
  __syncthreads();

  const int wave = threadIdx.x >> 5;
  const int lane = threadIdx.x & 31;
  const int n = lane & 15;
  const int mb = (lane >> 4) << 3;
  const bf16_t* W1e = W1t + (size_t)e * FFD * DD;   // [FF, D] n-major
  const float*  b1e = b1 + (size_t)e * FFD;

  // ---- FC1 + ReLU: each wave owns 512 of the 4096 hidden cols ----
  for (int c0 = wave * 512; c0 < wave * 512 + 512; c0 += 64) {
    v8f acc[4] = {};
    for (int kk = 0; kk < DD; kk += 32) {
      __builtin_prefetch(W1e + (size_t)c0 * DD + kk + 64, 0, 1);
      v16bf a = load_a_bf16(aT + kk, DD);
#pragma unroll
      for (int j = 0; j < 4; ++j) {
        v16bf b = load_bt_bf16(W1e + (size_t)(c0 + j * 16) * DD + kk, DD);
        acc[j] = wmma_bf16(a, b, acc[j]);
      }
    }
#pragma unroll
    for (int j = 0; j < 4; ++j)
#pragma unroll
      for (int r = 0; r < 8; ++r) {
        float hv = acc[j][r] + b1e[c0 + j * 16 + n];
        hT[(mb + r) * FFD + c0 + j * 16 + n] = (bf16_t)fmaxf(hv, 0.f);
      }
  }
  __syncthreads();

  // ---- FC2 + weighted scatter-add: each wave owns 128 of the 1024 cols ----
  const bf16_t* W2e = W2t + (size_t)e * DD * FFD;   // [D, FF] n-major
  const float*  b2e = b2 + (size_t)e * DD;
  for (int c0 = wave * 128; c0 < wave * 128 + 128; c0 += 64) {
    v8f acc[4] = {};
    for (int kk = 0; kk < FFD; kk += 32) {
      __builtin_prefetch(W2e + (size_t)c0 * FFD + kk + 64, 0, 1);
      v16bf a = load_a_bf16(hT + kk, FFD);
#pragma unroll
      for (int j = 0; j < 4; ++j) {
        v16bf b = load_bt_bf16(W2e + (size_t)(c0 + j * 16) * FFD + kk, FFD);
        acc[j] = wmma_bf16(a, b, acc[j]);
      }
    }
#pragma unroll
    for (int j = 0; j < 4; ++j)
#pragma unroll
      for (int r = 0; r < 8; ++r) {
        int tk = stok[mb + r];
        if (tk >= 0) {
          float val = (acc[j][r] + b2e[c0 + j * 16 + n]) * swt[mb + r];
          atomicAdd(&moe[(size_t)tk * DD + c0 + j * 16 + n], val);
        }
      }
  }
}

// ---------------------------------------------------------------------------
// Host launcher
// ---------------------------------------------------------------------------
extern "C" void kernel_launch(void* const* d_in, const int* in_sizes, int n_in,
                              void* d_out, int out_size, void* d_ws, size_t ws_size,
                              hipStream_t stream) {
  const float* x    = (const float*)d_in[0];
  const float* Wq   = (const float*)d_in[1];
  const float* Wk   = (const float*)d_in[2];
  const float* Wv   = (const float*)d_in[3];
  const float* gW   = (const float*)d_in[4];
  const float* W1   = (const float*)d_in[5];
  const float* b1   = (const float*)d_in[6];
  const float* W2   = (const float*)d_in[7];
  const float* b2   = (const float*)d_in[8];
  const float* ln1g = (const float*)d_in[9];
  const float* ln1b = (const float*)d_in[10];
  const float* ln2g = (const float*)d_in[11];
  const float* ln2b = (const float*)d_in[12];
  float* out = (float*)d_out;

  char* ws = (char*)d_ws;
  size_t off = 0;
  auto alloc = [&](size_t bytes) {
    char* p = ws + off;
    off += (bytes + 255) & ~(size_t)255;
    return p;
  };
  const size_t QS = (size_t)BB * HH * TT * HS;   // 2M elements
  const size_t NS = (size_t)BB * TT * DD;        // 2M elements
  // f32 intermediates
  float* attn = (float*)alloc(NS * 4);
  float* y    = (float*)alloc(NS * 4);
  float* moe  = (float*)alloc(NS * 4);
  // bf16 activations
  bf16_t* xbf = (bf16_t*)alloc(NS * 2);
  bf16_t* ybf = (bf16_t*)alloc(NS * 2);
  bf16_t* qb  = (bf16_t*)alloc(QS * 2);
  bf16_t* kb  = (bf16_t*)alloc(QS * 2);
  bf16_t* vtb = (bf16_t*)alloc(QS * 2);
  // bf16 transposed weights (L2-resident working set)
  bf16_t* Wqt = (bf16_t*)alloc((size_t)HH * DD * HS * 2);
  bf16_t* Wkt = (bf16_t*)alloc((size_t)HH * DD * HS * 2);
  bf16_t* Wvt = (bf16_t*)alloc((size_t)HH * DD * HS * 2);
  bf16_t* W1t = (bf16_t*)alloc((size_t)EE * DD * FFD * 2);
  bf16_t* W2t = (bf16_t*)alloc((size_t)EE * FFD * DD * 2);
  // routing state
  int*   counts   = (int*)alloc(EE * 4);
  int*   tok_list = (int*)alloc((size_t)EE * NTOK * 4);
  float* wt_list  = (float*)alloc((size_t)EE * NTOK * 4);

  // 0) precision prep: bf16 conversion (+ transpose for all GEMM B operands)
  cvt_bf16_kernel<<<(int)(NS / 4 / 256), 256, 0, stream>>>(x, xbf, (int)(NS / 4));
  tcvt_kernel<<<dim3(DD / 32, HS / 32, HH), 256, 0, stream>>>(Wq, Wqt, DD, HS);
  tcvt_kernel<<<dim3(DD / 32, HS / 32, HH), 256, 0, stream>>>(Wk, Wkt, DD, HS);
  tcvt_kernel<<<dim3(DD / 32, HS / 32, HH), 256, 0, stream>>>(Wv, Wvt, DD, HS);
  tcvt_kernel<<<dim3(DD / 32, FFD / 32, EE), 256, 0, stream>>>(W1, W1t, DD, FFD);
  tcvt_kernel<<<dim3(FFD / 32, DD / 32, EE), 256, 0, stream>>>(W2, W2t, FFD, DD);

  // 1) QKV projections (bf16 WMMA; v written transposed)
  qkv_kernel<<<dim3(NTOK / 64, HH, 3), 128, 0, stream>>>(
      xbf, Wqt, Wkt, Wvt, qb, kb, vtb);
  // 2) causal flash attention
  attn_kernel<<<dim3(TT / 64, BB * HH), 128, 0, stream>>>(qb, kb, vtb, attn);
  // 3) y = x + LN(attn)   (also emit ybf for the MoE GEMMs)
  add_ln_kernel<<<NTOK, 256, 0, stream>>>(x, attn, ln1g, ln1b, y, ybf);
  // 4) zero routing state + moe accumulator
  (void)hipMemsetAsync(counts, 0, EE * 4, stream);
  (void)hipMemsetAsync(moe, 0, NS * 4, stream);
  // 5) top-2 gate + token bucketing
  gate_kernel<<<NTOK, 256, 0, stream>>>(y, gW, counts, tok_list, wt_list);
  // 6) routed grouped expert GEMM
  moe_kernel<<<dim3(NTOK / 16, EE), 256, 0, stream>>>(
      ybf, W1t, b1, W2t, b2, counts, tok_list, wt_list, moe);
  // 7) out = y + LN(moe)
  add_ln_kernel<<<NTOK, 256, 0, stream>>>(y, moe, ln2g, ln2b, out, nullptr);
}
